// CustomCLIP_61624190763009
// MI455X (gfx1250) — matrix operand order repaired and checked by reading, not compiled, not run on hardware
//
#include <hip/hip_runtime.h>
#include <hip/hip_bf16.h>

// ---------------------------------------------------------------------------
// CustomCLIP fused forward for gfx1250 (MI455X), wave32 + WMMA bf16.
//   b=32, m=196 (pad 208), d=512, n=51 (pad 64), c=1000, TOPK=50, TEMP=5
// ---------------------------------------------------------------------------

typedef __attribute__((ext_vector_type(16))) __bf16 v16bf;
typedef __attribute__((ext_vector_type(8)))  float  v8f;

union FragBF { v16bf v; uint4 u[2]; };

#define M_PAD 208
#define N_PAD 64
#define SIM_STRIDE 65          // N_PAD + 1 : LDS bank-conflict padding
#define NEGF  (-3.0e38f)

__device__ __forceinline__ unsigned short f2bf(float f) {
    unsigned int u = __float_as_uint(f);
    u = (u + 0x7FFFu + ((u >> 16) & 1u)) >> 16;   // round-to-nearest-even
    return (unsigned short)u;
}
__device__ __forceinline__ float bf2f(unsigned short h) {
    return __uint_as_float(((unsigned int)h) << 16);
}

// --------------------------- row L2-normalize (fp32 -> fp32), width 512 -----
__global__ __launch_bounds__(256)
void row_norm_f32(const float* __restrict__ src, float* __restrict__ dst) {
    __shared__ float red[8];
    const int row = blockIdx.x, tid = threadIdx.x;
    const float* sp = src + (size_t)row * 512;
    float x0 = sp[tid], x1 = sp[tid + 256];
    float ss = x0 * x0 + x1 * x1;
    #pragma unroll
    for (int off = 16; off > 0; off >>= 1) ss += __shfl_xor(ss, off, 32);
    if ((tid & 31) == 0) red[tid >> 5] = ss;
    __syncthreads();
    if (tid == 0) {
        float s = 0.f;
        #pragma unroll
        for (int i = 0; i < 8; ++i) s += red[i];
        red[0] = rsqrtf(s);
    }
    __syncthreads();
    const float inv = red[0];
    float* dp = dst + (size_t)row * 512;
    dp[tid] = x0 * inv; dp[tid + 256] = x1 * inv;
}

// ------------------- row L2-normalize + pack bf16 with padded rows ----------
// mode 0: loc  rows = 32*208 ; row r: b=r/208, m=r%208 ; valid m<196
// mode 1: txt  rows = 1000*64; row r: c=r/64,  n=r%64  ; valid n<51 (src (n,c,d))
__global__ __launch_bounds__(256)
void norm_pack_kernel(const float* __restrict__ src, unsigned short* __restrict__ dst,
                      int mode) {
    __shared__ float red[8];
    const int r = blockIdx.x, tid = threadIdx.x;
    unsigned short* dp = dst + (size_t)r * 512;
    const float* sp = nullptr;
    if (mode == 0) {
        int b = r / M_PAD, m = r % M_PAD;
        if (m < 196) sp = src + ((size_t)b * 196 + m) * 512;
    } else {
        int c = r / N_PAD, n = r % N_PAD;
        if (n < 51)  sp = src + ((size_t)n * 1000 + c) * 512;
    }
    if (sp == nullptr) {                    // pad row -> zeros (block-uniform)
        dp[tid] = 0; dp[tid + 256] = 0;
        return;
    }
    float x0 = sp[tid], x1 = sp[tid + 256];
    float ss = x0 * x0 + x1 * x1;
    #pragma unroll
    for (int off = 16; off > 0; off >>= 1) ss += __shfl_xor(ss, off, 32);
    if ((tid & 31) == 0) red[tid >> 5] = ss;
    __syncthreads();
    if (tid == 0) {
        float s = 0.f;
        #pragma unroll
        for (int i = 0; i < 8; ++i) s += red[i];
        red[0] = rsqrtf(s);
    }
    __syncthreads();
    const float inv = red[0];
    dp[tid]       = f2bf(x0 * inv);
    dp[tid + 256] = f2bf(x1 * inv);
}

// ----------------- mean over n of normalized text: meanraw[c][d] ------------
__global__ __launch_bounds__(256)
void mean_kernel(const unsigned short* __restrict__ txtb, float* __restrict__ meanraw) {
    const int idx = blockIdx.x * 256 + threadIdx.x;      // 1000*512 total
    if (idx >= 1000 * 512) return;
    const int c = idx >> 9, d = idx & 511;
    const unsigned short* p = txtb + ((size_t)c * N_PAD) * 512 + d;
    float s = 0.f;
    for (int n = 0; n < 51; ++n) s += bf2f(p[(size_t)n * 512]);
    meanraw[idx] = s * (1.0f / 51.0f);
}

// ------------- v[c][n] = softmax_n( 5 * mean_txt[c] . txt[n,c] ) ------------
__global__ __launch_bounds__(64)
void v_kernel(const float* __restrict__ meann, const unsigned short* __restrict__ txtb,
              float* __restrict__ vmat) {
    __shared__ float lg[64];
    __shared__ float ssum;
    const int c = blockIdx.x, t = threadIdx.x;
    float dot = 0.f;
    if (t < 51) {
        const float* mp = meann + (size_t)c * 512;
        const unsigned short* tp = txtb + ((size_t)c * N_PAD + t) * 512;
        for (int d = 0; d < 512; ++d) dot += mp[d] * bf2f(tp[d]);
    }
    lg[t] = (t < 51) ? dot * 5.0f : NEGF;
    __syncthreads();
    if (t == 0) {
        float mx = lg[0];
        for (int i = 1; i < 51; ++i) mx = fmaxf(mx, lg[i]);
        float s = 0.f;
        for (int i = 0; i < 51; ++i) { lg[i] = __expf(lg[i] - mx); s += lg[i]; }
        ssum = s;
    }
    __syncthreads();
    vmat[(size_t)c * N_PAD + t] = (t < 51) ? (lg[t] / ssum) : 0.0f;
}

// -------- w_raw[b][m] = img_n[b] . loc_n[b][m]   (one wave per row) ---------
__global__ __launch_bounds__(256)
void wraw_kernel(const float* __restrict__ imgn, const unsigned short* __restrict__ locb,
                 float* __restrict__ wraw) {
    const int gw   = (blockIdx.x * 256 + threadIdx.x) >> 5;
    const int lane = threadIdx.x & 31;
    if (gw >= 32 * M_PAD) return;
    const int b = gw / M_PAD;
    const float* ip = imgn + (size_t)b * 512;
    const unsigned short* lp = locb + (size_t)gw * 512;
    float s = 0.f;
    for (int d = lane; d < 512; d += 32) s += ip[d] * bf2f(lp[d]);
    #pragma unroll
    for (int off = 16; off > 0; off >>= 1) s += __shfl_xor(s, off, 32);
    if (lane == 0) wraw[gw] = s;
}

// -------- base[b][c] = img_n[b] . mean_txt[c]    (one wave per dot) ---------
__global__ __launch_bounds__(256)
void base_kernel(const float* __restrict__ imgn, const float* __restrict__ meann,
                 float* __restrict__ base) {
    const int gw   = (blockIdx.x * 256 + threadIdx.x) >> 5;
    const int lane = threadIdx.x & 31;
    if (gw >= 32 * 1000) return;
    const int b = gw / 1000, c = gw % 1000;
    const float* ip = imgn + (size_t)b * 512;
    const float* mp = meann + (size_t)c * 512;
    float s = 0.f;
    for (int d = lane; d < 512; d += 32) s += ip[d] * mp[d];
    #pragma unroll
    for (int off = 16; off > 0; off >>= 1) s += __shfl_xor(s, off, 32);
    if (lane == 0) base[gw] = s;
}

// ---------------------------------------------------------------------------
// Main fused kernel: one block per (b,c).
//  Phase 1: simT[m][n] = loc_n[b,m] . txt_n[c,n] via v_wmma_f32_16x16x32_bf16.
//   A = loc (m rows), B = txt (n cols). Wave w owns m-tiles {w, w+8} (13 total)
//   and iterates all 4 n-tiles with persistent accumulators, so each A/B
//   fragment is reused: 12 loads feed 8 WMMAs (1.5 loads/WMMA vs 4 before).
//  Phase 2: per n: top-50 over m by iterative wave argmax; softmax over the
//   gathered w_raw values; accumulate v[c,n] * S2/S1.
// ---------------------------------------------------------------------------
__global__ __launch_bounds__(256)
void clip_main_kernel(const unsigned short* __restrict__ locb,  // [32][208][512]
                      const unsigned short* __restrict__ txtb,  // [1000][64][512]
                      const float* __restrict__ vmat,           // [1000][64]
                      const float* __restrict__ wraw,           // [32][208]
                      const float* __restrict__ base,           // [32][1000]
                      const float* __restrict__ lscale,
                      float* __restrict__ out) {                // [32][1000]
    __shared__ float simT[M_PAD][SIM_STRIDE];   // [m][n], padded stride
    __shared__ float bias_acc;

    const int blk  = blockIdx.x;
    const int b    = blk / 1000;
    const int c    = blk % 1000;
    const int tid  = threadIdx.x;
    const int wave = tid >> 5;
    const int lane = tid & 31;
    const int hi   = lane >> 4;       // half-wave select per WMMA layout
    const int l15  = lane & 15;

    if (tid == 0) bias_acc = 0.0f;

    const unsigned short* locp = locb + (size_t)b * (M_PAD * 512);
    const unsigned short* txtp = txtb + (size_t)c * (N_PAD * 512);

    // ---- Phase 1 ----
    const int tm0 = wave;                 // first m-tile
    const int tm1 = wave + 8;             // second m-tile (waves 0..4 only)
    const bool two = (tm1 < 13);          // wave-uniform

    v8f acc0[4], acc1[4];
    #pragma unroll
    for (int tn = 0; tn < 4; ++tn) { acc0[tn] = (v8f){}; acc1[tn] = (v8f){}; }

    // A rows: lane holds M=l15; K slots [k0+hi*8 .. +7] and [k0+16+hi*8 .. +7]
    const unsigned short* a0row = locp + (size_t)(tm0 * 16 + l15) * 512;
    const unsigned short* a1row = locp + (size_t)((two ? tm1 : tm0) * 16 + l15) * 512;
    // B cols: lane holds N=l15; K slots [k0+hi*8 .. +7] and [k0+16+hi*8 .. +7]
    const unsigned short* brow  = txtp + (size_t)l15 * 512;

    for (int k0 = 0; k0 < 512; k0 += 32) {
        FragBF a0, a1;
        a0.u[0] = *(const uint4*)(a0row + k0 + hi * 8);
        a0.u[1] = *(const uint4*)(a0row + k0 + 16 + hi * 8);
        if (two) {
            a1.u[0] = *(const uint4*)(a1row + k0 + hi * 8);
            a1.u[1] = *(const uint4*)(a1row + k0 + 16 + hi * 8);
        }
        #pragma unroll
        for (int tn = 0; tn < 4; ++tn) {
            FragBF bb;
            const unsigned short* bp = brow + (size_t)tn * (16 * 512) + k0 + hi * 8;
            bb.u[0] = *(const uint4*)(bp);
            bb.u[1] = *(const uint4*)(bp + 16);
            acc0[tn] = __builtin_amdgcn_wmma_f32_16x16x32_bf16(
                           false, a0.v, false, bb.v, (short)0, acc0[tn], false, false);
            if (two)
                acc1[tn] = __builtin_amdgcn_wmma_f32_16x16x32_bf16(
                               false, a1.v, false, bb.v, (short)0, acc1[tn], false, false);
        }
    }
    // D: VGPR r -> row M = tm*16 + r + hi*8, col N = tn*16 + l15
    #pragma unroll
    for (int tn = 0; tn < 4; ++tn) {
        const int ncol = tn * 16 + l15;
        #pragma unroll
        for (int r = 0; r < 8; ++r)
            simT[tm0 * 16 + r + hi * 8][ncol] = acc0[tn][r];
        if (two) {
            #pragma unroll
            for (int r = 0; r < 8; ++r)
                simT[tm1 * 16 + r + hi * 8][ncol] = acc1[tn][r];
        }
    }
    __syncthreads();

    // ---- Phase 2: top-50 + weighted softmax reduction ----
    float wave_sum = 0.f;
    for (int n = wave; n < 51; n += 8) {
        float vals[7];
        #pragma unroll
        for (int j = 0; j < 7; ++j) {
            const int m = lane + 32 * j;
            vals[j] = (m < 196) ? simT[m][n] : NEGF;
        }
        float S1 = 0.f, S2 = 0.f;
        for (int it = 0; it < 50; ++it) {
            float bv = vals[0]; int bj = 0;
            #pragma unroll
            for (int j = 1; j < 7; ++j)
                if (vals[j] > bv) { bv = vals[j]; bj = j; }
            int bm = lane + 32 * bj;
            #pragma unroll
            for (int off = 16; off > 0; off >>= 1) {
                const float ov = __shfl_xor(bv, off, 32);
                const int   om = __shfl_xor(bm, off, 32);
                if (ov > bv || (ov == bv && om < bm)) { bv = ov; bm = om; }
            }
            const float e = __expf(5.0f * wraw[b * M_PAD + bm]); // uniform addr
            S1 += e;
            S2 += bv * e;
            if ((bm & 31) == lane) vals[bm >> 5] = NEGF;         // invalidate
        }
        wave_sum += vmat[(size_t)c * N_PAD + n] * (S2 / S1);
    }
    if (lane == 0) atomicAdd(&bias_acc, wave_sum);
    __syncthreads();

    if (tid == 0)
        out[b * 1000 + c] = __expf(lscale[0]) * (base[b * 1000 + c] + bias_acc);
}

// ---------------------------------------------------------------------------
extern "C" void kernel_launch(void* const* d_in, const int* in_sizes, int n_in,
                              void* d_out, int out_size, void* d_ws, size_t ws_size,
                              hipStream_t stream) {
    const float* img_raw = (const float*)d_in[0];   // (32,512)
    const float* loc_raw = (const float*)d_in[1];   // (32,196,512)
    const float* txt_raw = (const float*)d_in[2];   // (51,1000,512)
    const float* lscale  = (const float*)d_in[3];   // (1,)
    float* out = (float*)d_out;                     // (32,1000)

    char* ws = (char*)d_ws;
    size_t off = 0;
    auto take = [&](size_t bytes) -> char* {
        char* p = ws + off;
        off += (bytes + 255) & ~(size_t)255;
        return p;
    };
    float*          img_n   = (float*)         take((size_t)32 * 512 * 4);
    unsigned short* locb    = (unsigned short*)take((size_t)32 * M_PAD * 512 * 2);
    unsigned short* txtb    = (unsigned short*)take((size_t)1000 * N_PAD * 512 * 2);
    float*          meanraw = (float*)         take((size_t)1000 * 512 * 4);
    float*          meann   = (float*)         take((size_t)1000 * 512 * 4);
    float*          vmat    = (float*)         take((size_t)1000 * N_PAD * 4);
    float*          wrawp   = (float*)         take((size_t)32 * M_PAD * 4);
    float*          basep   = (float*)         take((size_t)32 * 1000 * 4);

    row_norm_f32   <<<32,            256, 0, stream>>>(img_raw, img_n);
    norm_pack_kernel<<<32 * M_PAD,   256, 0, stream>>>(loc_raw, locb, 0);
    norm_pack_kernel<<<1000 * N_PAD, 256, 0, stream>>>(txt_raw, txtb, 1);
    mean_kernel    <<<(1000 * 512) / 256, 256, 0, stream>>>(txtb, meanraw);
    row_norm_f32   <<<1000,          256, 0, stream>>>(meanraw, meann);
    v_kernel       <<<1000,           64, 0, stream>>>(meann, txtb, vmat);
    wraw_kernel    <<<(32 * M_PAD * 32) / 256, 256, 0, stream>>>(img_n, locb, wrawp);
    base_kernel    <<<(32 * 1000 * 32) / 256,  256, 0, stream>>>(img_n, meann, basep);
    clip_main_kernel<<<32 * 1000,    256, 0, stream>>>(locb, txtb, vmat, wrawp,
                                                       basep, lscale, out);
}
